// SigLIPKMoeHead_77970836291863
// MI455X (gfx1250) — compile-verified, matrix operand order checked by reading, not run on hardware
//
#include <hip/hip_runtime.h>
#include <hip/hip_bf16.h>
#include <math.h>

typedef __bf16 bf16;
typedef __attribute__((ext_vector_type(16))) __bf16 v16bf;
typedef __attribute__((ext_vector_type(8)))  __bf16 v8bf;
typedef __attribute__((ext_vector_type(8)))  float  v8f;

static constexpr int B_  = 64;
static constexpr int S_  = 1024;
static constexpr int D_  = 768;
static constexpr int T_  = 4;
static constexpr int E_  = 8;
static constexpr int H_  = 3072;
static constexpr int NH_ = 12;
static constexpr int HD_ = 64;
static constexpr int NTOK = B_ * T_;          // 256

// ---- workspace layout (bytes) ----
static constexpr size_t OFF_KV   = 0;                                          // bf16 [B*S][1536] K|V
static constexpr size_t OFF_Q    = OFF_KV   + (size_t)B_*S_*2*D_*sizeof(bf16); // f32 [T][768] (pre-scaled)
static constexpr size_t OFF_CTX  = OFF_Q    + (size_t)T_*D_*4;                 // f32 [256][768]
static constexpr size_t OFF_RES  = OFF_CTX  + (size_t)NTOK*D_*4;               // f32 [256][768]
static constexpr size_t OFF_X    = OFF_RES  + (size_t)NTOK*D_*4;               // f32 [256][768] (post-LN)
static constexpr size_t OFF_H    = OFF_X    + (size_t)NTOK*D_*4;               // f32 [256][E*H]
static constexpr size_t OFF_Y    = OFF_H    + (size_t)NTOK*E_*H_*4;            // f32 [256][E*D]
static constexpr size_t OFF_COMB = OFF_Y    + (size_t)NTOK*E_*D_*4;            // f32 [256][E]
static constexpr size_t OFF_ID   = OFF_COMB + (size_t)NTOK*E_*4;               // f32 [16] imp|disp

// ---- output layout (floats) ----
static constexpr size_t OUT_MAIN = 0;                  // [256][768]
static constexpr size_t OUT_LOSS = (size_t)NTOK*D_;    // [1]
static constexpr size_t OUT_ATTN = OUT_LOSS + 1;       // [256][1024]

// =====================================================================
// Generic WMMA GEMM:  C[M,N] = A[M,K] * B[N,K]^T  (+bias, +epilogue)
// fp32 operands, converted to bf16 while staging into LDS.
// BK = 64 (two WMMA K-steps per stage), register double-buffered:
// next global tile is loaded into registers while current tile computes.
// MODE 0: kv-proj   -> bf16 store
// MODE 1: fc1+gelu  -> f32 store
// MODE 2: fc2 (per-expert via blockIdx.z) -> f32 store
// =====================================================================
static constexpr int LDSK = 72;   // 144B row stride: 16B-aligned, conflict-free

template <int MODE>
__global__ __launch_bounds__(256)
void k_gemm(const float* __restrict__ A, int lda,
            const float* __restrict__ Bw, int ldb,
            const float* __restrict__ bias,
            void* __restrict__ Cout, int ldc, int K)
{
  int zofs = 0;
  if (MODE == 2) {
    const int e = blockIdx.z;
    A    += (size_t)e * H_;                 // h rows stride lda=E*H, expert block at e*H
    Bw   += (size_t)e * D_ * (size_t)ldb;   // fc2_w[e]
    bias += e * D_;
    zofs  = e * D_;                         // column offset into y row
  }

  __shared__ bf16 As[128][LDSK];
  __shared__ bf16 Bs[128][LDSK];

  const int tid  = threadIdx.x;
  const int m0   = blockIdx.y * 128;
  const int n0   = blockIdx.x * 128;

  const int w    = tid >> 5;          // wave 0..7
  const int lane = tid & 31;
  const int wRow = (w >> 1) * 32;     // 0,32,64,96
  const int wCol = (w & 1) * 64;      // 0,64
  const int lrow = lane & 15;         // M (A) / N (B) within 16-tile
  const int kh   = (lane >> 4) * 8;   // K half select per ISA layout

  v8f acc[2][4];
#pragma unroll
  for (int mi = 0; mi < 2; ++mi)
#pragma unroll
    for (int nj = 0; nj < 4; ++nj)
#pragma unroll
      for (int r = 0; r < 8; ++r) acc[mi][nj][r] = 0.0f;

  // loader geometry: 16 threads per row (64 cols / float4), 16 rows/pass, 8 passes
  const int arow = tid >> 4;          // 0..15
  const int acol = (tid & 15) * 4;    // 0..60

  const float* Ald = A  + (size_t)(m0 + arow) * lda + acol;
  const float* Bld = Bw + (size_t)(n0 + arow) * ldb + acol;

  float4 ra[8], rb[8];
  // prefetch first tile (k0 = 0)
#pragma unroll
  for (int p = 0; p < 8; ++p) {
    ra[p] = *(const float4*)(Ald + (size_t)(p * 16) * lda);
    rb[p] = *(const float4*)(Bld + (size_t)(p * 16) * ldb);
  }

  for (int k0 = 0; k0 < K; k0 += 64) {
    // commit prefetched registers to LDS
#pragma unroll
    for (int p = 0; p < 8; ++p) {
      bf16* da = &As[arow + p * 16][acol];
      da[0] = (bf16)ra[p].x; da[1] = (bf16)ra[p].y; da[2] = (bf16)ra[p].z; da[3] = (bf16)ra[p].w;
      bf16* db = &Bs[arow + p * 16][acol];
      db[0] = (bf16)rb[p].x; db[1] = (bf16)rb[p].y; db[2] = (bf16)rb[p].z; db[3] = (bf16)rb[p].w;
    }
    __syncthreads();

    // prefetch next tile into registers (overlaps with WMMA below)
    if (k0 + 64 < K) {
#pragma unroll
      for (int p = 0; p < 8; ++p) {
        ra[p] = *(const float4*)(Ald + (size_t)(p * 16) * lda + (k0 + 64));
        rb[p] = *(const float4*)(Bld + (size_t)(p * 16) * ldb + (k0 + 64));
      }
    }

    // two WMMA K-steps from this stage
#pragma unroll
    for (int kk = 0; kk < 64; kk += 32) {
      v16bf av[2], bv[4];
#pragma unroll
      for (int mi = 0; mi < 2; ++mi) {
        v8bf p0 = *(const v8bf*)&As[wRow + mi * 16 + lrow][kk + kh];
        v8bf p1 = *(const v8bf*)&As[wRow + mi * 16 + lrow][kk + kh + 16];
        av[mi] = __builtin_shufflevector(p0, p1, 0,1,2,3,4,5,6,7,8,9,10,11,12,13,14,15);
      }
#pragma unroll
      for (int nj = 0; nj < 4; ++nj) {
        v8bf p0 = *(const v8bf*)&Bs[wCol + nj * 16 + lrow][kk + kh];
        v8bf p1 = *(const v8bf*)&Bs[wCol + nj * 16 + lrow][kk + kh + 16];
        bv[nj] = __builtin_shufflevector(p0, p1, 0,1,2,3,4,5,6,7,8,9,10,11,12,13,14,15);
      }
#pragma unroll
      for (int mi = 0; mi < 2; ++mi)
#pragma unroll
        for (int nj = 0; nj < 4; ++nj)
          acc[mi][nj] = __builtin_amdgcn_wmma_f32_16x16x32_bf16(
              false, av[mi], false, bv[nj], (short)0, acc[mi][nj], false, false);
    }
    __syncthreads();
  }

  const int rofs = (lane >> 4) * 8;   // C/D: lanes 16..31 hold M = r+8
#pragma unroll
  for (int mi = 0; mi < 2; ++mi)
#pragma unroll
    for (int nj = 0; nj < 4; ++nj)
#pragma unroll
      for (int r = 0; r < 8; ++r) {
        const int gm = m0 + wRow + mi * 16 + r + rofs;
        const int gn = n0 + wCol + nj * 16 + lrow;
        float v = acc[mi][nj][r] + bias[gn];
        if (MODE == 1) {   // gelu (tanh approx)
          const float u = v;
          v = 0.5f * u * (1.0f + tanhf(0.7978845608028654f * (u + 0.044715f * u * u * u)));
        }
        if (MODE == 0)
          ((bf16*)Cout)[(size_t)gm * ldc + gn] = (bf16)v;
        else
          ((float*)Cout)[(size_t)gm * ldc + zofs + gn] = v;
      }
}

// =====================================================================
// q = (probe @ wq^T + bq) * (1/sqrt(HD)) ; T*D outputs, 12 blocks
// =====================================================================
__global__ __launch_bounds__(256)
void k_qproj(const float* __restrict__ probe, const float* __restrict__ Wq,
             const float* __restrict__ bq, float* __restrict__ q)
{
  const int gid = blockIdx.x * 256 + threadIdx.x;   // 0..3071
  const int t = gid / D_, d = gid % D_;
  float a = bq[d];
  const float* p = probe + (size_t)t * D_;
  const float* wv = Wq + (size_t)d * D_;
  for (int k = 0; k < D_; ++k) a += p[k] * wv[k];
  q[gid] = a * 0.125f;  // 1/sqrt(64)
}

// =====================================================================
// Attention per (b,t): scores -> softmax -> attn mean + ctx
// =====================================================================
__global__ __launch_bounds__(256)
void k_attn(const bf16* __restrict__ kv, const float* __restrict__ q,
            float* __restrict__ ctx, float* __restrict__ attn_out)
{
  __shared__ float sc[NH_][S_];   // 48KB
  __shared__ float qs[D_];
  __shared__ float red[256];
  __shared__ float inv[NH_];

  const int b = blockIdx.x, t = blockIdx.y, tid = threadIdx.x;

  for (int i = tid; i < D_; i += 256) qs[i] = q[t * D_ + i];
  __syncthreads();

  const bf16* kvb = kv + (size_t)b * S_ * (2 * D_);

  for (int i = tid; i < NH_ * S_; i += 256) {
    const int h = i % NH_, s = i / NH_;
    const v8bf* kp = (const v8bf*)(kvb + (size_t)s * (2 * D_) + h * HD_);
    float a = 0.f;
#pragma unroll
    for (int c = 0; c < 8; ++c) {
      v8bf kk = kp[c];
#pragma unroll
      for (int j = 0; j < 8; ++j) a += qs[h * HD_ + c * 8 + j] * (float)kk[j];
    }
    sc[h][s] = a;
  }
  __syncthreads();

  for (int h = 0; h < NH_; ++h) {
    float m = -3.0e38f;
    for (int s = tid; s < S_; s += 256) m = fmaxf(m, sc[h][s]);
    red[tid] = m; __syncthreads();
    for (int o = 128; o > 0; o >>= 1) { if (tid < o) red[tid] = fmaxf(red[tid], red[tid + o]); __syncthreads(); }
    const float mx = red[0]; __syncthreads();
    float sm = 0.f;
    for (int s = tid; s < S_; s += 256) { float e = __expf(sc[h][s] - mx); sc[h][s] = e; sm += e; }
    red[tid] = sm; __syncthreads();
    for (int o = 128; o > 0; o >>= 1) { if (tid < o) red[tid] += red[tid + o]; __syncthreads(); }
    if (tid == 0) inv[h] = 1.0f / red[0];
    __syncthreads();
  }

  float* aw = attn_out + (size_t)(b * T_ + t) * S_;
  for (int s = tid; s < S_; s += 256) {
    float a = 0.f;
#pragma unroll
    for (int h = 0; h < NH_; ++h) a += sc[h][s] * inv[h];
    aw[s] = a * (1.0f / NH_);
  }

  float* cx = ctx + (size_t)(b * T_ + t) * D_;
  for (int i = tid; i < D_; i += 256) {
    const int h = i >> 6, d = i & 63;
    float a = 0.f;
    const bf16* vp = kvb + D_ + h * HD_ + d;
    for (int s = 0; s < S_; ++s) a += sc[h][s] * (float)vp[(size_t)s * (2 * D_)];
    cx[i] = a * inv[h];
  }
}

// =====================================================================
// out-proj per token: res = ctx @ Wo^T + bo
// =====================================================================
__global__ __launch_bounds__(256)
void k_outproj(const float* __restrict__ ctx, const float* __restrict__ W,
               const float* __restrict__ bias, float* __restrict__ res)
{
  __shared__ float xs[D_];
  const int tok = blockIdx.x, tid = threadIdx.x;
  const float* c = ctx + (size_t)tok * D_;
  for (int i = tid; i < D_; i += 256) xs[i] = c[i];
  __syncthreads();
  for (int n = tid; n < D_; n += 256) {
    float a = bias[n];
    const float* wr = W + (size_t)n * D_;
    for (int d = 0; d < D_; ++d) a += xs[d] * wr[d];
    res[(size_t)tok * D_ + n] = a;
  }
}

// =====================================================================
// LayerNorm + gate softmax + top-2 + importance/dispatch atomics
// =====================================================================
__global__ __launch_bounds__(256)
void k_ln_gate(const float* __restrict__ res, const float* __restrict__ lnw,
               const float* __restrict__ lnb, const float* __restrict__ gate_w,
               float* __restrict__ x, float* __restrict__ comb,
               float* __restrict__ impdisp)
{
  __shared__ float xs[D_];
  __shared__ float red[256];
  __shared__ float le[E_];
  __shared__ float stats[2];
  const int tok = blockIdx.x, tid = threadIdx.x;
  const int t = tok % T_;
  const float* rrow = res + (size_t)tok * D_;

  float s = 0.f;
  for (int i = tid; i < D_; i += 256) { float v = rrow[i]; xs[i] = v; s += v; }
  red[tid] = s; __syncthreads();
  for (int o = 128; o > 0; o >>= 1) { if (tid < o) red[tid] += red[tid + o]; __syncthreads(); }
  if (tid == 0) stats[0] = red[0] * (1.0f / D_);
  __syncthreads();
  const float mu = stats[0];
  float vs = 0.f;
  for (int i = tid; i < D_; i += 256) { float d = xs[i] - mu; vs += d * d; }
  red[tid] = vs; __syncthreads();
  for (int o = 128; o > 0; o >>= 1) { if (tid < o) red[tid] += red[tid + o]; __syncthreads(); }
  if (tid == 0) stats[1] = rsqrtf(red[0] * (1.0f / D_) + 1e-6f);
  __syncthreads();
  const float rs = stats[1];
  for (int i = tid; i < D_; i += 256) {
    float v = (xs[i] - mu) * rs * lnw[i] + lnb[i];
    xs[i] = v;
    x[(size_t)tok * D_ + i] = v;
  }
  __syncthreads();

  if (tid < E_) {
    float a = 0.f;
    const float* g = gate_w + (size_t)t * D_ * E_ + tid;
    for (int d = 0; d < D_; ++d) a += xs[d] * g[(size_t)d * E_];
    le[tid] = a;
  }
  __syncthreads();

  if (tid == 0) {
    float mx = le[0];
    for (int e = 1; e < E_; ++e) mx = fmaxf(mx, le[e]);
    float ps[E_], sm = 0.f;
    for (int e = 0; e < E_; ++e) { ps[e] = __expf(le[e] - mx); sm += ps[e]; }
    const float is = 1.0f / sm;
    for (int e = 0; e < E_; ++e) ps[e] *= is;
    int i1 = 0;
    for (int e = 1; e < E_; ++e) if (ps[e] > ps[i1]) i1 = e;
    int i2 = (i1 == 0) ? 1 : 0;
    for (int e = 0; e < E_; ++e) if (e != i1 && ps[e] > ps[i2]) i2 = e;
    const float den = 1.0f / (ps[i1] + ps[i2]);
    for (int e = 0; e < E_; ++e) {
      const float c = (e == i1) ? ps[i1] * den : ((e == i2) ? ps[i2] * den : 0.f);
      comb[(size_t)tok * E_ + e] = c;
      atomicAdd(&impdisp[e], ps[e]);
    }
    atomicAdd(&impdisp[E_ + i1], 1.0f);
    atomicAdd(&impdisp[E_ + i2], 1.0f);
  }
}

__global__ void k_zero(float* p, int n)
{
  const int i = blockIdx.x * blockDim.x + threadIdx.x;
  if (i < n) p[i] = 0.f;
}

__global__ __launch_bounds__(256)
void k_combine(const float* __restrict__ res, const float* __restrict__ y,
               const float* __restrict__ comb, float* __restrict__ outp)
{
  __shared__ float cw[E_];
  const int tok = blockIdx.x, tid = threadIdx.x;
  if (tid < E_) cw[tid] = comb[(size_t)tok * E_ + tid];
  __syncthreads();
  for (int i = tid; i < D_; i += 256) {
    float a = res[(size_t)tok * D_ + i];
    const float* yr = y + (size_t)tok * E_ * D_ + i;
#pragma unroll
    for (int e = 0; e < E_; ++e) a += cw[e] * yr[(size_t)e * D_];
    outp[(size_t)tok * D_ + i] = a;
  }
}

__global__ void k_loss(const float* __restrict__ impdisp, float* __restrict__ outp)
{
  if (threadIdx.x == 0 && blockIdx.x == 0) {
    float s = 0.f;
    for (int e = 0; e < E_; ++e)
      s += (impdisp[e] * (1.0f / NTOK)) * (impdisp[E_ + e] * (1.0f / NTOK));
    outp[0] = (float)E_ * s;
  }
}

// =====================================================================
extern "C" void kernel_launch(void* const* d_in, const int* in_sizes, int n_in,
                              void* d_out, int out_size, void* d_ws, size_t ws_size,
                              hipStream_t stream)
{
  const float* hidden = (const float*)d_in[0];
  const float* probe  = (const float*)d_in[1];
  const float* ipw    = (const float*)d_in[2];
  const float* ipb    = (const float*)d_in[3];
  const float* opw    = (const float*)d_in[4];
  const float* opb    = (const float*)d_in[5];
  const float* lnw    = (const float*)d_in[6];
  const float* lnb    = (const float*)d_in[7];
  const float* gatew  = (const float*)d_in[8];
  const float* fc1w   = (const float*)d_in[9];
  const float* fc1b   = (const float*)d_in[10];
  const float* fc2w   = (const float*)d_in[11];
  const float* fc2b   = (const float*)d_in[12];
  float* outp = (float*)d_out;

  char* ws = (char*)d_ws;
  bf16*  kvbuf = (bf16*)(ws + OFF_KV);
  float* qbuf  = (float*)(ws + OFF_Q);
  float* ctxb  = (float*)(ws + OFF_CTX);
  float* resb  = (float*)(ws + OFF_RES);
  float* xbuf  = (float*)(ws + OFF_X);
  float* hbuf  = (float*)(ws + OFF_H);
  float* ybuf  = (float*)(ws + OFF_Y);
  float* combb = (float*)(ws + OFF_COMB);
  float* idb   = (float*)(ws + OFF_ID);

  // q projection (probe is broadcast over batch -> only T rows)
  k_qproj<<<dim3(12), dim3(256), 0, stream>>>(probe, ipw, ipb, qbuf);

  // K/V projection: [65536 x 1536] = hidden[65536 x 768] @ (wk|wv)[1536 x 768]^T
  k_gemm<0><<<dim3(12, 512), dim3(256), 0, stream>>>(
      hidden, D_, ipw + (size_t)D_ * D_, D_, ipb + D_, kvbuf, 2 * D_, D_);

  // attention per (b, t)
  k_attn<<<dim3(B_, T_), dim3(256), 0, stream>>>(kvbuf, qbuf, ctxb, outp + OUT_ATTN);

  // out-proj -> residual
  k_outproj<<<dim3(NTOK), dim3(256), 0, stream>>>(ctxb, opw, opb, resb);

  // zero importance/dispatch accumulators
  k_zero<<<dim3(1), dim3(32), 0, stream>>>(idb, 16);

  // LN + gate + top-2
  k_ln_gate<<<dim3(NTOK), dim3(256), 0, stream>>>(resb, lnw, lnb, gatew, xbuf, combb, idb);

  // fc1 (dense over all experts): [256 x 24576] = x[256 x 768] @ fc1_w[24576 x 768]^T, gelu
  k_gemm<1><<<dim3(192, 2), dim3(256), 0, stream>>>(
      xbuf, D_, fc1w, D_, fc1b, hbuf, E_ * H_, D_);

  // fc2 per expert: y[256 x E*768] ; A = h (lda = E*H), B = fc2_w[e] (768 x 3072)
  k_gemm<2><<<dim3(6, 2, E_), dim3(256), 0, stream>>>(
      hbuf, E_ * H_, fc2w, H_, fc2b, ybuf, E_ * D_, H_);

  // residual + combine, and moe loss
  k_combine<<<dim3(NTOK), dim3(256), 0, stream>>>(resb, ybuf, combb, outp);
  k_loss<<<dim3(1), dim3(1), 0, stream>>>(idb, outp + OUT_LOSS);
}
